// PointerLayer_8967891714319
// MI455X (gfx1250) — matrix operand order repaired and checked by reading, not compiled, not run on hardware
//
#include <hip/hip_runtime.h>

#define NB 4
#define SEQ 1024
#define HID 1024
#define HEADS 12
#define HEAD_SIZE 64
#define OUT_DIM (HEADS * HEAD_SIZE * 2)   // 1536
#define INFV 1000000000000.0f

#define LDSX (128 * 72)   // X chunk: 128 rows x 64 (+8 pad) halfs
#define LDSW (64 * 72)    // W chunk: 64 rows x 64 (+8 pad) halfs

typedef __attribute__((ext_vector_type(16))) _Float16 v16h;
typedef __attribute__((ext_vector_type(8)))  _Float16 v8h;
typedef __attribute__((ext_vector_type(4)))  _Float16 v4h;
typedef __attribute__((ext_vector_type(8)))  float    v8f;
typedef __attribute__((ext_vector_type(4)))  float    v4f;

// ---- WMMA fragment loaders (layouts per CDNA5 ISA 7.12.2, wave32) ----
// A: 16x32 f16, row-major source with leading dim `ld` (f16 elements).
// lane 0-15: M=lane, K = {0..7} in halves 0..7, {16..23} in halves 8..15;
// lane 16-31: same M, K ranges shifted by +8.
__device__ __forceinline__ v16h frag_a(const _Float16* base, int ld, int row0,
                                       int kk, int lane) {
  const int m = lane & 15, khi = lane >> 4;
  const _Float16* p = base + (size_t)(row0 + m) * ld + kk + 8 * khi;
  v8h lo = *(const v8h*)p;
  v8h hi = *(const v8h*)(p + 16);
  v16h a;
#pragma unroll
  for (int t = 0; t < 8; ++t) { a[t] = lo[t]; a[8 + t] = hi[t]; }
  return a;
}

// B: 32x16 f16. Source stored "N-major rows of K" (row n holds K contiguous),
// leading dim `ld`. lane 0-15: N=lane, K = {0..7},{16..23} ; lane 16-31: N=lane-16, K shifted by +8.
__device__ __forceinline__ v16h frag_b(const _Float16* base, int ld, int row0,
                                       int kk, int lane) {
  const int n = lane & 15, kh = lane >> 4;
  const _Float16* p = base + (size_t)(row0 + n) * ld + kk + 8 * kh;
  v8h lo = *(const v8h*)p;
  v8h hi = *(const v8h*)(p + 16);
  v16h b;
#pragma unroll
  for (int t = 0; t < 8; ++t) { b[t] = lo[t]; b[8 + t] = hi[t]; }
  return b;
}

__device__ __forceinline__ v8f wmma_f16(v16h a, v16h b, v8f c) {
  return __builtin_amdgcn_wmma_f32_16x16x32_f16(false, a, false, b, (short)0, c,
                                                false, false);
}

// =====================================================================
// Kernel 1: proj = X @ W^T + b, then RoPE, scatter q/k as f16 [b][h][n][d]
// Block: 256 threads = 8 waves; block tile = 128 rows x 64 cols.
// Double-buffered LDS (2 x 27 KB): global loads for chunk k+1 issue before
// the WMMAs of chunk k; one barrier per chunk.
// Grid: (4096/128) * (1536/64) = 32 * 24 = 768 blocks.
// =====================================================================
__global__ __launch_bounds__(256) void proj_rope_kernel(
    const float* __restrict__ X,    // [4096,1024]
    const float* __restrict__ W,    // [1536,1024]
    const float* __restrict__ bias, // [1536]
    _Float16* __restrict__ qbuf,    // [4,12,1024,64]
    _Float16* __restrict__ kbuf) {  // [4,12,1024,64]
  __shared__ _Float16 Xl[2][LDSX];
  __shared__ _Float16 Wl[2][LDSW];

  const int tid  = threadIdx.x;
  const int lane = tid & 31;
  const int wv   = tid >> 5;               // wave 0..7 -> 16-row slice
  const int mg   = blockIdx.x % 32;        // row group (128 rows)
  const int ng   = blockIdx.x / 32;        // col group (64 cols)
  const int row_base = mg * 128;
  const int col_base = ng * 64;

  // Per-thread staging addresses: thread -> (row = tid/16 (+16*i), col4 = (tid&15)*4)
  const int srow = tid >> 4;
  const int sc4  = (tid & 15) * 4;
  const float* xp = X + (size_t)(row_base + srow) * HID + sc4;  // + i*16*HID, + kb
  const float* wp = W + (size_t)(col_base + srow) * HID + sc4;  // + i*16*HID, + kb
  _Float16* xls = &Xl[0][srow * 72 + sc4];                      // + i*16*72
  _Float16* wls = &Wl[0][srow * 72 + sc4];

  v8f acc[4] = {};

  // ---- preload chunk 0 ----
  {
    v4f xr[8], wr[4];
#pragma unroll
    for (int i = 0; i < 8; ++i) xr[i] = *(const v4f*)(xp + (size_t)i * 16 * HID);
#pragma unroll
    for (int i = 0; i < 4; ++i) wr[i] = *(const v4f*)(wp + (size_t)i * 16 * HID);
#pragma unroll
    for (int i = 0; i < 8; ++i) {
      v4h hv = {(_Float16)xr[i].x, (_Float16)xr[i].y, (_Float16)xr[i].z, (_Float16)xr[i].w};
      *(v4h*)(xls + i * 16 * 72) = hv;
    }
#pragma unroll
    for (int i = 0; i < 4; ++i) {
      v4h hv = {(_Float16)wr[i].x, (_Float16)wr[i].y, (_Float16)wr[i].z, (_Float16)wr[i].w};
      *(v4h*)(wls + i * 16 * 72) = hv;
    }
  }
  __syncthreads();

  for (int kb = 0; kb < HID; kb += 64) {
    const int cur = (kb >> 6) & 1;
    const int nxt = cur ^ 1;
    const bool has_next = (kb + 64) < HID;

    // ---- issue global loads for the NEXT chunk first (hide latency) ----
    v4f xr[8], wr[4];
    if (has_next) {
      const float* xpn = xp + kb + 64;
      const float* wpn = wp + kb + 64;
#pragma unroll
      for (int i = 0; i < 8; ++i) xr[i] = *(const v4f*)(xpn + (size_t)i * 16 * HID);
#pragma unroll
      for (int i = 0; i < 4; ++i) wr[i] = *(const v4f*)(wpn + (size_t)i * 16 * HID);
    }

    // ---- compute on CURRENT chunk: preload all fragments, then 8 WMMAs ----
    {
      v16h a0 = frag_a(Xl[cur], 72, wv * 16, 0, lane);
      v16h a1 = frag_a(Xl[cur], 72, wv * 16, 32, lane);
      v16h b0[4], b1[4];
#pragma unroll
      for (int t = 0; t < 4; ++t) b0[t] = frag_b(Wl[cur], 72, t * 16, 0, lane);
#pragma unroll
      for (int t = 0; t < 4; ++t) b1[t] = frag_b(Wl[cur], 72, t * 16, 32, lane);
#pragma unroll
      for (int t = 0; t < 4; ++t) acc[t] = wmma_f16(a0, b0[t], acc[t]);
#pragma unroll
      for (int t = 0; t < 4; ++t) acc[t] = wmma_f16(a1, b1[t], acc[t]);
    }

    // ---- convert + store NEXT chunk into the other LDS buffer ----
    if (has_next) {
      _Float16* xd = xls + nxt * 0 + (nxt ? (LDSX) : 0) - (cur ? LDSX : 0) + (cur ? LDSX : 0);
      // (use explicit indexing instead of pointer juggling)
      (void)xd;
#pragma unroll
      for (int i = 0; i < 8; ++i) {
        v4h hv = {(_Float16)xr[i].x, (_Float16)xr[i].y, (_Float16)xr[i].z, (_Float16)xr[i].w};
        *(v4h*)(&Xl[nxt][srow * 72 + sc4] + i * 16 * 72) = hv;
      }
#pragma unroll
      for (int i = 0; i < 4; ++i) {
        v4h hv = {(_Float16)wr[i].x, (_Float16)wr[i].y, (_Float16)wr[i].z, (_Float16)wr[i].w};
        *(v4h*)(&Wl[nxt][srow * 72 + sc4] + i * 16 * 72) = hv;
      }
    }
    __syncthreads();
  }

  // ---- epilogue: bias + RoPE + scatter to q/k (f16, [b][h][n][d]) ----
  const int khi   = lane >> 4;
  const int ln15  = lane & 15;
  const int gm0   = row_base + wv * 16;      // global proj row base
  const int b_idx = gm0 / SEQ;               // 128-row block never crosses batch

#pragma unroll
  for (int t = 0; t < 4; ++t) {
    int o    = col_base + t * 16 + ln15;     // proj column for this lane
    int hh   = o >> 7;                       // head
    int t128 = o & 127;
    int isk  = (t128 >= 64);
    int d    = t128 & 63;
    int j    = d >> 1;
    // inv_freq = 10000^(-2j/64) = exp(-j * 2*ln(1e4)/64)
    float inv = __expf(-0.28782313662425572f * (float)j);
    float bo  = bias[o];
    float sgn = (o & 1) ? 1.0f : -1.0f;      // even col: -x_odd*sin ; odd: +x_even*sin
    _Float16* outp =
        (isk ? kbuf : qbuf) + ((size_t)(b_idx * HEADS + hh) * SEQ) * HEAD_SIZE + d;
#pragma unroll
    for (int r = 0; r < 8; ++r) {
      int   npos = (gm0 + r + 8 * khi) & (SEQ - 1);
      float val  = acc[t][r] + bo;
      float part = __shfl_xor(val, 1, 32);   // paired column lives in adjacent lane
      float ang  = (float)npos * inv;
      float s = __sinf(ang), c = __cosf(ang);
      float ro = val * c + sgn * part * s;
      outp[(size_t)npos * HEAD_SIZE] = (_Float16)ro;
    }
  }
}

// =====================================================================
// Kernel 2: logits[b,h,m,n] = q . k, masked + causal + scaled, fp32 out.
// Block: 256 threads = 8 waves; wave owns 16 rows, sweeps all 1024 cols.
// K-fragments are software-pipelined one 16-col tile ahead.
// Grid: (1024/128, 4*12) = (8, 48).
// =====================================================================
__global__ __launch_bounds__(256) void attn_logits_kernel(
    const _Float16* __restrict__ qbuf,
    const _Float16* __restrict__ kbuf,
    const float* __restrict__ mask,   // [4,1024]
    float* __restrict__ out) {        // [4,12,1024,1024]
  const int tid   = threadIdx.x;
  const int lane  = tid & 31;
  const int wv    = tid >> 5;
  const int bh    = blockIdx.y;            // 0..47
  const int b_idx = bh / HEADS;
  const int m0    = blockIdx.x * 128 + wv * 16;
  const int khi   = lane >> 4;
  const int ln15  = lane & 15;

  const _Float16* Q = qbuf + (size_t)bh * SEQ * HEAD_SIZE;
  const _Float16* K = kbuf + (size_t)bh * SEQ * HEAD_SIZE;
  float* O = out + (size_t)bh * SEQ * SEQ;

  // Q fragments for this wave's 16 rows (K-dim 64 -> two k-steps), loaded once.
  v16h a0 = frag_a(Q, HEAD_SIZE, m0, 0, lane);
  v16h a1 = frag_a(Q, HEAD_SIZE, m0, 32, lane);

  float mq[8];
#pragma unroll
  for (int r = 0; r < 8; ++r) mq[r] = mask[b_idx * SEQ + m0 + r + 8 * khi];

  // Pipeline: fetch tile n0=0 fragments before the loop.
  v16h b0 = frag_b(K, HEAD_SIZE, 0, 0, lane);
  v16h b1 = frag_b(K, HEAD_SIZE, 0, 32, lane);

  for (int n0 = 0; n0 < SEQ; n0 += 16) {
    // Prefetch next tile's fragments (clamped; redundant on last iter).
    const int np = (n0 + 16 < SEQ) ? (n0 + 16) : n0;
    v16h nb0 = frag_b(K, HEAD_SIZE, np, 0, lane);
    v16h nb1 = frag_b(K, HEAD_SIZE, np, 32, lane);

    v8f c = {};
    c = wmma_f16(a0, b0, c);
    c = wmma_f16(a1, b1, c);

    int   n  = n0 + ln15;
    float mk = mask[b_idx * SEQ + n];
#pragma unroll
    for (int r = 0; r < 8; ++r) {
      int   m = m0 + r + 8 * khi;
      float v = c[r];
      v = v * mq[r] + (1.0f - mq[r]) * (-INFV);
      v = v * mk    + (1.0f - mk)    * (-INFV);
      v -= (n < m) ? INFV : 0.0f;              // tril(.., -1) * INF
      // 201 MB output stream: bypass caches (keep q/k resident in L2)
      __builtin_nontemporal_store(v * 0.125f, &O[(size_t)m * SEQ + n]);
    }
    b0 = nb0;
    b1 = nb1;
  }
}

extern "C" void kernel_launch(void* const* d_in, const int* in_sizes, int n_in,
                              void* d_out, int out_size, void* d_ws, size_t ws_size,
                              hipStream_t stream) {
  const float* X    = (const float*)d_in[0];   // context_output [4,1024,1024]
  const float* mask = (const float*)d_in[1];   // [4,1024]
  const float* W    = (const float*)d_in[2];   // [1536,1024]
  const float* bias = (const float*)d_in[3];   // [1536]
  float* out = (float*)d_out;                  // [4,12,1024,1024]

  _Float16* qbuf = (_Float16*)d_ws;                               // 6 MB (f16)
  _Float16* kbuf = qbuf + (size_t)NB * HEADS * SEQ * HEAD_SIZE;   // 6 MB (f16)

  proj_rope_kernel<<<dim3(32 * 24), 256, 0, stream>>>(X, W, bias, qbuf, kbuf);
  attn_logits_kernel<<<dim3(8, 48), 256, 0, stream>>>(qbuf, kbuf, mask, out);
}
